// Gemma4FullAttention_24936580120984
// MI455X (gfx1250) — matrix operand order repaired and checked by reading, not compiled, not run on hardware
//
#include <hip/hip_runtime.h>
#include <hip/hip_bf16.h>
#include <math.h>

typedef __attribute__((ext_vector_type(16))) _Float16 v16h;
typedef __attribute__((ext_vector_type(8)))  _Float16 v8h;
typedef __attribute__((ext_vector_type(4)))  _Float16 v4h;
typedef __attribute__((ext_vector_type(8)))  float    v8f;
typedef __attribute__((ext_vector_type(4)))  int      v4i;

// ---------------- problem constants ----------------
constexpr int Bsz = 2;
constexpr int Ls  = 2048;
constexpr int HSn = 2048;
constexpr int NHn = 16;
constexpr int NKVn = 4;
constexpr int HDn = 128;
constexpr float SCALEc = 0.08838834764831845f;  // 128^-0.5

// ---------------- CDNA5 async global->LDS staging helpers ----------------
#if defined(__has_builtin)
#if __has_builtin(__builtin_amdgcn_global_load_async_to_lds_b128)
#define HAVE_ASYNC_LDS 1
#endif
#endif

__device__ inline void wait_async0() {
#if defined(__has_builtin) && __has_builtin(__builtin_amdgcn_s_wait_asynccnt)
  __builtin_amdgcn_s_wait_asynccnt(0);
#else
  asm volatile("s_wait_asynccnt 0" ::: "memory");
#endif
}

__device__ inline void wait_ds0() {
  asm volatile("s_wait_dscnt 0" ::: "memory");
}

// ---------------- fragment helpers (contiguous k order, not the ISA table order) ----------------
// A-matrix 16x32 f16 fragment from a row-major tile, leading dim ld.
// lane(0-15): M=lane, K={0..15}; lane(16-31): M=lane-16, K={16..31}
// One contiguous 16-element k-run -> two 16B vector loads.
__device__ inline v16h load_a16(const _Float16* __restrict__ t, int ld, int hl, int hi) {
  const _Float16* row = t + hl * ld + hi * 16;
  v8h lo = *(const v8h*)(row);
  v8h hh = *(const v8h*)(row + 8);
  return __builtin_shufflevector(lo, hh, 0, 1, 2, 3, 4, 5, 6, 7,
                                         8, 9, 10, 11, 12, 13, 14, 15);
}

// ---------------- generic f32-in/f32-out GEMM via f16 WMMA ----------------
// C[M,N] = A[M,K] * B[K,N]; M%128==0, N%128==0, K%32==0
#define BM 128
#define BN 128
#define BKK 32

__global__ __launch_bounds__(256)
void gemm_wmma_kernel(const float* __restrict__ A, const float* __restrict__ B,
                      float* __restrict__ C, int M, int N, int K) {
  // double-buffered tiles: stage t+1 while WMMAs consume t
  __shared__ __align__(32) _Float16 As[2][BM * BKK];    // row-major [m][k]
  __shared__ __align__(32) _Float16 BsT[2][BN * BKK];   // transposed [n][k]
  const int tid  = threadIdx.x;
  const int wave = tid >> 5;
  const int lane = tid & 31;
  const int hl = lane & 15;
  const int hi = (lane >> 4) & 1;
  const int bm = blockIdx.y * BM;
  const int bn = blockIdx.x * BN;
  const int wr = wave >> 1;   // 0..3 -> 32-row strip
  const int wc = wave & 1;    // 0..1 -> 64-col strip

  // stage tile at k-offset kk into buffer bufi: batch all global loads first
  // (one clause, one wait), then convert+store; branchless clamped prefetch.
  auto stage_tile = [&](int bufi, int kk) {
    int kpf = (kk + BKK < K) ? (kk + BKK) : kk;  // uniform select, no branch
    float4 fa[4], fb[4];
    #pragma unroll
    for (int it = 0; it < 4; ++it) {
      int i4 = tid * 4 + it * 1024;
      int ra = i4 >> 5, ca = i4 & 31;
      int rb = i4 >> 7, cb = i4 & 127;
      fa[it] = *(const float4*)&A[(size_t)(bm + ra) * K + kk + ca];
      fb[it] = *(const float4*)&B[(size_t)(kk + rb) * N + bn + cb];
      __builtin_prefetch(&A[(size_t)(bm + ra) * K + kpf + ca], 0, 3);
      __builtin_prefetch(&B[(size_t)(kpf + rb) * N + bn + cb], 0, 3);
    }
    #pragma unroll
    for (int it = 0; it < 4; ++it) {
      int i4 = tid * 4 + it * 1024;
      int rb = i4 >> 7, cb = i4 & 127;
      v4h h4 = {(_Float16)fa[it].x, (_Float16)fa[it].y,
                (_Float16)fa[it].z, (_Float16)fa[it].w};
      *(v4h*)&As[bufi][i4] = h4;
      BsT[bufi][(cb + 0) * BKK + rb] = (_Float16)fb[it].x;
      BsT[bufi][(cb + 1) * BKK + rb] = (_Float16)fb[it].y;
      BsT[bufi][(cb + 2) * BKK + rb] = (_Float16)fb[it].z;
      BsT[bufi][(cb + 3) * BKK + rb] = (_Float16)fb[it].w;
    }
  };

  v8f acc[2][4] = {};
  const int nk = K / BKK;
  stage_tile(0, 0);

  for (int t = 0; t < nk; ++t) {
    __syncthreads();                       // buffer t&1 ready; buffer (t+1)&1 free
    if (t + 1 < nk) stage_tile((t + 1) & 1, (t + 1) * BKK);

    const _Float16* Ab = As[t & 1];
    const _Float16* Bb = BsT[t & 1];
    v16h af[2], bf[4];
    #pragma unroll
    for (int u = 0; u < 2; ++u)
      af[u] = load_a16(&Ab[(wr * 32 + u * 16) * BKK], BKK, hl, hi);
    #pragma unroll
    for (int u = 0; u < 4; ++u)   // contiguous 32B per lane: k-run hi*16..+15
      bf[u] = *(const v16h*)&Bb[(wc * 64 + u * 16 + hl) * BKK + hi * 16];

    #pragma unroll
    for (int i = 0; i < 2; ++i)
      #pragma unroll
      for (int jj = 0; jj < 4; ++jj)
        acc[i][jj] = __builtin_amdgcn_wmma_f32_16x16x32_f16(
            false, af[i], false, bf[jj], (short)0, acc[i][jj], false, false);
  }

  #pragma unroll
  for (int i = 0; i < 2; ++i) {
    int row0 = bm + wr * 32 + i * 16 + hi * 8;
    #pragma unroll
    for (int jj = 0; jj < 4; ++jj) {
      int col = bn + wc * 64 + jj * 16 + hl;
      #pragma unroll
      for (int v = 0; v < 8; ++v)
        C[(size_t)(row0 + v) * N + col] = acc[i][jj][v];
    }
  }
}

// ---------------- RMSNorm + RoPE + f16 transpose ----------------
// src: [rows, nheads*128] f32 (rows = b*Ls + l)
// transpose_out==0: dst [b, head, l, 128] f16 ; ==1: dst [b, head, 128, L] f16 (V^T)
__global__ __launch_bounds__(256)
void normrope_kernel(const float* __restrict__ src, const float* __restrict__ w,
                     _Float16* __restrict__ dst, int rows, int nheads, int dorope,
                     int transpose_out) {
  int wid  = (blockIdx.x * blockDim.x + threadIdx.x) >> 5;
  int lane = threadIdx.x & 31;
  int h = wid % nheads;
  int r = wid / nheads;
  if (r >= rows) return;
  int l  = r % Ls;
  int bI = r / Ls;
  const float* x = src + (size_t)r * (nheads * HDn) + h * HDn;

  float x0 = x[lane], x1 = x[lane + 32], x2 = x[lane + 64], x3 = x[lane + 96];

  if (w) {
    float ss = x0 * x0 + x1 * x1 + x2 * x2 + x3 * x3;
    #pragma unroll
    for (int m = 16; m >= 1; m >>= 1) ss += __shfl_xor(ss, m, 32);
    float inv = rsqrtf(ss * (1.0f / 128.0f) + 1e-6f);
    x0 = x0 * inv * w[lane];      x1 = x1 * inv * w[lane + 32];
    x2 = x2 * inv * w[lane + 64]; x3 = x3 * inv * w[lane + 96];
  }
  if (dorope) {
    const float LOG2_BASE = 13.287712379549449f; // log2(10000)
    float invf0 = exp2f(-(2.0f * lane        / 128.0f) * LOG2_BASE);
    float invf1 = exp2f(-(2.0f * (lane + 32) / 128.0f) * LOG2_BASE);
    float c0, s0, c1, s1;
    __sincosf((float)l * invf0, &s0, &c0);
    __sincosf((float)l * invf1, &s1, &c1);
    float o0 = x0 * c0 - x2 * s0;
    float o2 = x2 * c0 + x0 * s0;
    float o1 = x1 * c1 - x3 * s1;
    float o3 = x3 * c1 + x1 * s1;
    x0 = o0; x1 = o1; x2 = o2; x3 = o3;
  }
  if (transpose_out) {
    _Float16* yt = dst + ((size_t)(bI * nheads + h) * HDn) * Ls + l;
    yt[(size_t)(lane)      * Ls] = (_Float16)x0;
    yt[(size_t)(lane + 32) * Ls] = (_Float16)x1;
    yt[(size_t)(lane + 64) * Ls] = (_Float16)x2;
    yt[(size_t)(lane + 96) * Ls] = (_Float16)x3;
  } else {
    _Float16* y = dst + (((size_t)bI * nheads + h) * Ls + l) * HDn;
    y[lane] = (_Float16)x0;      y[lane + 32] = (_Float16)x1;
    y[lane + 64] = (_Float16)x2; y[lane + 96] = (_Float16)x3;
  }
}

// 16B-granule global->LDS copy: async on CDNA5 toolchains that expose it.
// Builtin signature (from clang diagnostics): (AS1 v4i* src, AS3 v4i* dst, Imm, Imm)
__device__ inline void stage16(const _Float16* g, _Float16* l) {
#if defined(HAVE_ASYNC_LDS)
  __builtin_amdgcn_global_load_async_to_lds_b128(
      (__attribute__((address_space(1))) v4i*)(void*)(const void*)g,
      (__attribute__((address_space(3))) v4i*)(void*)l, 0, 0);
#else
  *(uint4*)l = *(const uint4*)g;
#endif
}

__device__ inline void stage_wait() {
#if defined(HAVE_ASYNC_LDS)
  wait_async0();
#else
  wait_ds0();
#endif
}

// ---------------- flash attention (per wave: 16 query rows) ----------------
// Qh: [B,NH,L,128] f16, Kh: [B,NKV,L,128] f16, VhT: [B,NKV,128,L] f16,
// Out: [B,L,NH*128] f32
#define ATT_WAVES 4
#define KV_TILE (32 * 128)  // f16 elems (K tile 32x128, later reused as V^T 128x32)
#define P_TILE  (16 * 32)

__global__ __launch_bounds__(128)
void attn_kernel(const _Float16* __restrict__ Qh, const _Float16* __restrict__ Kh,
                 const _Float16* __restrict__ VhT, float* __restrict__ Out) {
  __shared__ __align__(32) _Float16 smem[ATT_WAVES * (KV_TILE + P_TILE)];
  const int wave = threadIdx.x >> 5;
  const int lane = threadIdx.x & 31;
  const int hl = lane & 15;
  const int hi = (lane >> 4) & 1;
  const int h = blockIdx.y;
  const int b = blockIdx.z;
  const int qbase = (blockIdx.x * ATT_WAVES + wave) * 16;
  const int j = h % NKVn;

  const _Float16* Q   = Qh  + ((size_t)(b * NHn  + h) * Ls) * HDn;
  const _Float16* Kp  = Kh  + ((size_t)(b * NKVn + j) * Ls) * HDn;
  const _Float16* VpT = VhT + ((size_t)(b * NKVn + j) * HDn) * Ls;
  _Float16* KVt = &smem[wave * (KV_TILE + P_TILE)];
  _Float16* Pt  = KVt + KV_TILE;

  // Q A-fragments over 4 HD chunks of 32 (contiguous global b128 loads)
  v16h qf[4];
  #pragma unroll
  for (int c = 0; c < 4; ++c)
    qf[c] = load_a16(Q + (size_t)qbase * HDn + c * 32, HDn, hl, hi);

  v8f o[8] = {};
  float mrow[8], lrow[8];
  #pragma unroll
  for (int i = 0; i < 8; ++i) { mrow[i] = -1e30f; lrow[i] = 0.0f; }

  const int nkb = (qbase + 15) / 32 + 1;  // causal: key blocks of 32
  for (int kb = 0; kb < nkb; ++kb) {
    const int kbase = kb * 32;

    // branchless clamped prefetch of next block's K / V^T rows into GL2
    {
      int kpf = (kb + 1 < nkb) ? kbase + 32 : kbase;  // uniform select
      __builtin_prefetch(Kp + (size_t)(kpf + lane) * HDn, 0, 3);
      __builtin_prefetch(VpT + (size_t)(lane * 4) * Ls + kpf, 0, 3);
    }

    // stage K tile (32 keys x 128) into per-wave LDS.
    // ASYNC ops are unordered with DS ops: ensure prior ds reads of this
    // buffer (last block's V^T fragments) are done before overwriting.
    wait_ds0();
    {
      const _Float16* g = Kp + (size_t)(kbase + lane) * HDn;
      _Float16* l = KVt + lane * HDn;
      #pragma unroll
      for (int u = 0; u < 16; ++u) stage16(g + u * 8, l + u * 8);
    }
    stage_wait();

    // S = Q * K^T : two 16x16 tiles, K-dim = 128 (4 WMMA each)
    v8f s0 = {}, s1 = {};
    #pragma unroll
    for (int c = 0; c < 4; ++c) {
      // K^T B-fragment: contiguous 32B run of one key row in LDS
      v16h bk0 = *(const v16h*)(KVt + (0 * 16 + hl) * HDn + c * 32 + hi * 16);
      v16h bk1 = *(const v16h*)(KVt + (16     + hl) * HDn + c * 32 + hi * 16);
      s0 = __builtin_amdgcn_wmma_f32_16x16x32_f16(false, qf[c], false, bk0, (short)0, s0, false, false);
      s1 = __builtin_amdgcn_wmma_f32_16x16x32_f16(false, qf[c], false, bk1, (short)0, s1, false, false);
    }

    // scale + causal mask; online softmax (rows live across the 16-lane half)
    float rmax[8];
    #pragma unroll
    for (int i = 0; i < 8; ++i) {
      int qi = qbase + i + 8 * hi;
      float v0 = s0[i] * SCALEc;
      float v1 = s1[i] * SCALEc;
      if (kbase + hl      > qi) v0 = -1e30f;
      if (kbase + 16 + hl > qi) v1 = -1e30f;
      s0[i] = v0; s1[i] = v1;
      float x = fmaxf(v0, v1);
      #pragma unroll
      for (int m = 8; m >= 1; m >>= 1) x = fmaxf(x, __shfl_xor(x, m, 32));
      rmax[i] = x;
    }
    float alpha[8];
    #pragma unroll
    for (int i = 0; i < 8; ++i) {
      float mn = fmaxf(mrow[i], rmax[i]);
      alpha[i] = __expf(mrow[i] - mn);
      mrow[i] = mn;
      float p0 = __expf(s0[i] - mn);
      float p1 = __expf(s1[i] - mn);
      s0[i] = p0; s1[i] = p1;
      float sum = p0 + p1;
      #pragma unroll
      for (int m = 8; m >= 1; m >>= 1) sum += __shfl_xor(sum, m, 32);
      lrow[i] = lrow[i] * alpha[i] + sum;
    }
    #pragma unroll
    for (int c = 0; c < 8; ++c)
      #pragma unroll
      for (int i = 0; i < 8; ++i) o[c][i] *= alpha[i];

    // transpose P (C-layout -> A-layout) through per-wave LDS
    #pragma unroll
    for (int i = 0; i < 8; ++i) {
      Pt[(i + 8 * hi) * 32 + hl]      = (_Float16)s0[i];
      Pt[(i + 8 * hi) * 32 + 16 + hl] = (_Float16)s1[i];
    }
    wait_ds0();
    v16h pf = load_a16(Pt, 32, hl, hi);

    // stage V^T tile (128 hd x 32 keys) into same LDS region.
    // Prior ds reads (K^T fragments, pf) must complete before async overwrite.
    wait_ds0();
    {
      #pragma unroll
      for (int u = 0; u < 4; ++u) {
        int hd = lane * 4 + u;
        const _Float16* g = VpT + (size_t)hd * Ls + kbase;
        _Float16* l = KVt + hd * 32;
        #pragma unroll
        for (int q4 = 0; q4 < 4; ++q4) stage16(g + q4 * 8, l + q4 * 8);
      }
    }
    stage_wait();

    // O += P * V : 8 column chunks; V^T fragment = contiguous 32B per lane
    #pragma unroll
    for (int c = 0; c < 8; ++c) {
      v16h vf = *(const v16h*)(KVt + (c * 16 + hl) * 32 + hi * 16);
      o[c] = __builtin_amdgcn_wmma_f32_16x16x32_f16(false, pf, false, vf, (short)0, o[c], false, false);
    }
  }

  // finalize & store: Out[b, q, h*128 + c*16 + hl]
  float invl[8];
  #pragma unroll
  for (int i = 0; i < 8; ++i) invl[i] = 1.0f / lrow[i];
  #pragma unroll
  for (int c = 0; c < 8; ++c) {
    #pragma unroll
    for (int i = 0; i < 8; ++i) {
      int q = qbase + i + 8 * hi;
      Out[((size_t)b * Ls + q) * (NHn * HDn) + h * HDn + c * 16 + hl] = o[c][i] * invl[i];
    }
  }
}

// ---------------- host launch ----------------
extern "C" void kernel_launch(void* const* d_in, const int* in_sizes, int n_in,
                              void* d_out, int out_size, void* d_ws, size_t ws_size,
                              hipStream_t stream) {
  const float* hidden = (const float*)d_in[0];
  const float* Wq = (const float*)d_in[1];
  const float* Wk = (const float*)d_in[2];
  const float* Wv = (const float*)d_in[3];
  const float* Wo = (const float*)d_in[4];
  const float* qnw = (const float*)d_in[5];
  const float* knw = (const float*)d_in[6];
  float* out = (float*)d_out;

  const size_t rows = (size_t)Bsz * Ls;                 // 4096
  char* ws = (char*)d_ws;
  size_t offQ  = 0;                                     // f32 Q  [4096,2048]
  size_t offK  = offQ  + rows * (NHn  * HDn) * 4;       // f32 K  [4096,512]
  size_t offV  = offK  + rows * (NKVn * HDn) * 4;       // f32 V  [4096,512]
  size_t offQh = offV  + rows * (NKVn * HDn) * 4;       // f16 Q head-major
  size_t offKh = offQh + rows * (NHn  * HDn) * 2;
  size_t offVh = offKh + rows * (NKVn * HDn) * 2;       // f16 V^T head-major

  float* Qf = (float*)(ws + offQ);
  float* Kf = (float*)(ws + offK);
  float* Vf = (float*)(ws + offV);
  _Float16* Qhp = (_Float16*)(ws + offQh);
  _Float16* Khp = (_Float16*)(ws + offKh);
  _Float16* Vhp = (_Float16*)(ws + offVh);
  float* AttO = Qf;  // reuse f32 Q buffer for attention output

  dim3 blk(256);
  // QKV projections
  gemm_wmma_kernel<<<dim3((NHn * HDn) / BN,  rows / BM), blk, 0, stream>>>(
      hidden, Wq, Qf, (int)rows, NHn * HDn, HSn);
  gemm_wmma_kernel<<<dim3((NKVn * HDn) / BN, rows / BM), blk, 0, stream>>>(
      hidden, Wk, Kf, (int)rows, NKVn * HDn, HSn);
  gemm_wmma_kernel<<<dim3((NKVn * HDn) / BN, rows / BM), blk, 0, stream>>>(
      hidden, Wv, Vf, (int)rows, NKVn * HDn, HSn);

  // RMSNorm + RoPE + transpose to head-major f16 (V also transposed to [hd][L])
  normrope_kernel<<<(unsigned)(rows * NHn  / 8), blk, 0, stream>>>(Qf, qnw, Qhp, (int)rows, NHn, 1, 0);
  normrope_kernel<<<(unsigned)(rows * NKVn / 8), blk, 0, stream>>>(Kf, knw, Khp, (int)rows, NKVn, 1, 0);
  normrope_kernel<<<(unsigned)(rows * NKVn / 8), blk, 0, stream>>>(Vf, nullptr, Vhp, (int)rows, NKVn, 0, 1);

  // flash attention: 1 wave per 16-query tile, 4 waves/block
  attn_kernel<<<dim3(Ls / (16 * ATT_WAVES), NHn, Bsz), dim3(128), 0, stream>>>(
      Qhp, Khp, Vhp, AttO);

  // output projection
  gemm_wmma_kernel<<<dim3(HSn / BN, rows / BM), blk, 0, stream>>>(
      AttO, Wo, out, (int)rows, HSn, NHn * HDn);
}
